// MambaBackbone_18777597018327
// MI455X (gfx1250) — matrix operand order repaired and checked by reading, not compiled, not run on hardware
//
#include <hip/hip_runtime.h>
#include <hip/hip_bf16.h>
#include <stdint.h>
#include <stddef.h>

#define D_MODEL  1024
#define N_BLOCKS 4
#define D_STATE  16
#define D_CONV   4
#define DT_RANK  64
#define D_INNER  2048
#define BATCH    2
#define SEQLEN   1024
#define ROWS     (BATCH * SEQLEN)        /* 2048 */
#define DBL_COLS (DT_RANK + 2 * D_STATE) /* 96 */

typedef __bf16 bf16_t;
typedef __attribute__((ext_vector_type(16))) __bf16 v16bf;
typedef __attribute__((ext_vector_type(8)))  __bf16 v8bf;
typedef __attribute__((ext_vector_type(8)))  float  v8f;

__device__ __forceinline__ bf16_t f2bf(float f) {
  uint32_t u = __builtin_bit_cast(uint32_t, f);
  uint32_t r = u + 0x7FFFu + ((u >> 16) & 1u);   // round-to-nearest-even
  uint16_t h = (uint16_t)(r >> 16);
  return __builtin_bit_cast(bf16_t, h);
}
__device__ __forceinline__ float siluf(float x)     { return x / (1.f + __expf(-x)); }
__device__ __forceinline__ float softplusf(float x) { return x > 20.f ? x : log1pf(__expf(x)); }

// ---------------------------------------------------------------------------
// Weight convert + transpose: W (K x N, f32, per-block) -> Wt (N x K, bf16)
// ---------------------------------------------------------------------------
__global__ void transpose_to_bf16(const float* __restrict__ W,
                                  bf16_t* __restrict__ Wt, int K, int N) {
  const size_t per = (size_t)K * N;
  const float*  Wb  = W  + per * blockIdx.y;
  bf16_t*       Wtb = Wt + per * blockIdx.y;
  for (size_t idx = (size_t)blockIdx.x * blockDim.x + threadIdx.x;
       idx < per; idx += (size_t)gridDim.x * blockDim.x) {
    int n = (int)(idx / K), k = (int)(idx % K);
    Wtb[idx] = f2bf(Wb[(size_t)k * N + n]);
  }
}

// ---------------------------------------------------------------------------
// f32 -> bf16, source rows strided (extract dbl[:, :DT_RANK])
// ---------------------------------------------------------------------------
__global__ void cvt_bf16_strided(const float* __restrict__ src, bf16_t* __restrict__ dst,
                                 int rows, int cols, int src_ld) {
  int idx = blockIdx.x * blockDim.x + threadIdx.x;
  if (idx >= rows * cols) return;
  int r = idx / cols, c = idx % cols;
  dst[idx] = f2bf(src[(size_t)r * src_ld + c]);
}

// ---------------------------------------------------------------------------
// LayerNorm over last dim D; optional f32 and/or bf16 outputs. 1 block / row.
// ---------------------------------------------------------------------------
__global__ void layernorm_kernel(const float* __restrict__ x,
                                 const float* __restrict__ w,
                                 const float* __restrict__ b,
                                 float* __restrict__ outf,
                                 bf16_t* __restrict__ outb, int D) {
  __shared__ float red[2][8];
  __shared__ float stats[2];
  const float* row = x + (size_t)blockIdx.x * D;
  float s = 0.f, ss = 0.f;
  for (int j = threadIdx.x; j < D; j += blockDim.x) {
    float v = row[j]; s += v; ss += v * v;
  }
  for (int off = 16; off > 0; off >>= 1) {
    s  += __shfl_down(s,  off, 32);
    ss += __shfl_down(ss, off, 32);
  }
  int wid = threadIdx.x >> 5, lane = threadIdx.x & 31;
  if (lane == 0) { red[0][wid] = s; red[1][wid] = ss; }
  __syncthreads();
  if (threadIdx.x == 0) {
    float ts = 0.f, tss = 0.f;
    for (int wv = 0; wv < 8; ++wv) { ts += red[0][wv]; tss += red[1][wv]; }
    float mean = ts / D;
    float var  = tss / D - mean * mean;
    stats[0] = mean;
    stats[1] = rsqrtf(var + 1e-5f);
  }
  __syncthreads();
  float mean = stats[0], rstd = stats[1];
  for (int j = threadIdx.x; j < D; j += blockDim.x) {
    float v = (row[j] - mean) * rstd * w[j] + b[j];
    if (outf) outf[(size_t)blockIdx.x * D + j] = v;
    if (outb) outb[(size_t)blockIdx.x * D + j] = f2bf(v);
  }
}

// ---------------------------------------------------------------------------
// Depthwise causal conv (K=4) + bias + SiLU on u-half of xz; writes f32 + bf16
// ---------------------------------------------------------------------------
__global__ void conv_silu_kernel(const float* __restrict__ xz,
                                 const float* __restrict__ cw,
                                 const float* __restrict__ cb,
                                 float* __restrict__ u, bf16_t* __restrict__ u_bf) {
  int idx = blockIdx.x * blockDim.x + threadIdx.x;       // [0, 2^22)
  int e  = idx & (D_INNER - 1);
  int l  = (idx >> 11) & (SEQLEN - 1);
  int bb = idx >> 21;
  float acc = cb[e];
#pragma unroll
  for (int k = 0; k < D_CONV; ++k) {
    int ls = l + k - (D_CONV - 1);
    if (ls >= 0)
      acc += xz[((size_t)(bb * SEQLEN + ls)) * (2 * D_INNER) + e] * cw[e * D_CONV + k];
  }
  float s = siluf(acc);
  size_t o = (size_t)(bb * SEQLEN + l) * D_INNER + e;
  u[o] = s;
  u_bf[o] = f2bf(s);
}

// ---------------------------------------------------------------------------
// WMMA GEMM, register-blocked: each wave owns a (16*TM) x (16*TN) output
// block. B fragments reused across TM WMMAs, A across TN; TM*TN independent
// accumulators keep the XDL pipe past the WMMA->WMMA hazard window.
//   C[MxN](f32) = A[MxK](bf16, row-major, lda) @ Bt[NxK](bf16, transposed)
// epilogue: mode 0 = store, 1 = accumulate (+=), 2 = softplus(c + bias[col])
// ---------------------------------------------------------------------------
__device__ __forceinline__ v16bf load_afrag(const bf16_t* __restrict__ arow,
                                            int kb, int khalf) {
  // A 16x32 bf16 layout (contiguous, matches B): lane<16 holds K = kb..kb+15,
  // lane>=16 holds K = kb+16..kb+31; row = lane&15.
  v8bf alo = *(const v8bf*)(arow + kb + khalf * 16);
  v8bf ahi = *(const v8bf*)(arow + kb + khalf * 16 + 8);
  v16bf a;
#pragma unroll
  for (int j = 0; j < 8; ++j) { a[j] = alo[j]; a[8 + j] = ahi[j]; }
  return a;
}

template <int TM, int TN>
__global__ void gemm_bf16_wmma(const bf16_t* __restrict__ A, int lda,
                               const bf16_t* __restrict__ Bt,
                               float* __restrict__ C, int ldc,
                               int M, int N, int K,
                               int mode, const float* __restrict__ bias) {
  const int lane  = threadIdx.x & 31;
  const int wave  = threadIdx.x >> 5;
  const int tile  = blockIdx.x * (blockDim.x >> 5) + wave;
  const int ntn   = N / (16 * TN);
  const int ntm   = M / (16 * TM);
  if (tile >= ntm * ntn) return;                // uniform per wave
  const int tm    = tile / ntn;
  const int tn    = tile % ntn;
  const int r     = lane & 15;
  const int khalf = lane >> 4;

  const bf16_t* arow[TM];
#pragma unroll
  for (int mi = 0; mi < TM; ++mi)
    arow[mi] = A + (size_t)(tm * 16 * TM + mi * 16 + r) * lda;
  const bf16_t* brow[TN];
#pragma unroll
  for (int ni = 0; ni < TN; ++ni)
    brow[ni] = Bt + (size_t)(tn * 16 * TN + ni * 16 + r) * K;

  v8f acc[TM][TN];
#pragma unroll
  for (int mi = 0; mi < TM; ++mi)
#pragma unroll
    for (int ni = 0; ni < TN; ++ni) acc[mi][ni] = (v8f){};

  const int ksteps = K >> 5;
  for (int kt = 0; kt < ksteps; ++kt) {
    const int kb = kt * 32;
    if (kt + 2 < ksteps) {                      // keep next lines warm (WGP scope)
#pragma unroll
      for (int mi = 0; mi < TM; ++mi) __builtin_prefetch(arow[mi] + kb + 64, 0, 3);
#pragma unroll
      for (int ni = 0; ni < TN; ++ni) __builtin_prefetch(brow[ni] + kb + 64, 0, 3);
    }
    // B 32x16: lane = column (n), lanes 0-15 hold K=kb..kb+15, lanes 16-31 +16
    v16bf bfrag[TN];
#pragma unroll
    for (int ni = 0; ni < TN; ++ni)
      bfrag[ni] = *(const v16bf*)(brow[ni] + kb + khalf * 16);
#pragma unroll
    for (int mi = 0; mi < TM; ++mi) {
      v16bf afrag = load_afrag(arow[mi], kb, khalf);
#pragma unroll
      for (int ni = 0; ni < TN; ++ni)
        acc[mi][ni] = __builtin_amdgcn_wmma_f32_16x16x32_bf16(
            /*neg_a=*/false, afrag, /*neg_b=*/false, bfrag[ni],
            /*c_mod=*/(short)0, acc[mi][ni],
            /*reuse_a=*/false, /*reuse_b=*/false);
    }
  }

  // D/C layout: VGPR v -> row base + khalf*8 + v, col base + (lane&15)
#pragma unroll
  for (int mi = 0; mi < TM; ++mi) {
#pragma unroll
    for (int ni = 0; ni < TN; ++ni) {
      const int col = tn * 16 * TN + ni * 16 + r;
#pragma unroll
      for (int v = 0; v < 8; ++v) {
        float val = acc[mi][ni][v];
        float* pc = C + (size_t)(tm * 16 * TM + mi * 16 + khalf * 8 + v) * ldc + col;
        if (mode == 1)      *pc += val;
        else if (mode == 2) *pc  = softplusf(val + bias[col]);
        else                *pc  = val;
      }
    }
  }
}

// ---------------------------------------------------------------------------
// Selective scan: 1 lane per (batch, channel). B/C for (b,l) fetched by one
// lane each (dbl cols 64..95) and broadcast across the wave via __shfl.
// ---------------------------------------------------------------------------
__global__ void scan_kernel(const float* __restrict__ dt,
                            const float* __restrict__ u,
                            const float* __restrict__ dbl,
                            const float* __restrict__ xz,
                            const float* __restrict__ A_log,
                            const float* __restrict__ skip_D,
                            bf16_t* __restrict__ y_bf) {
  int t = blockIdx.x * blockDim.x + threadIdx.x;   // [0, 4096)
  int b = t >> 11;
  int e = t & (D_INNER - 1);
  int lane = threadIdx.x & 31;

  float A[D_STATE], h[D_STATE];
#pragma unroll
  for (int n = 0; n < D_STATE; ++n) {
    A[n] = -__expf(A_log[(size_t)e * D_STATE + n]);
    h[n] = 0.f;
  }
  const float Dv = skip_D[e];

  for (int l = 0; l < SEQLEN; ++l) {
    const size_t row = (size_t)(b * SEQLEN + l);
    float dtv = dt[row * D_INNER + e];
    float uv  = u[row * D_INNER + e];
    float dtu = dtv * uv;
    // lane k reads dbl[row, 64+k]: k<16 => B[k], k>=16 => C[k-16]
    float bc  = dbl[row * DBL_COLS + DT_RANK + lane];
    float acc = 0.f;
#pragma unroll
    for (int n = 0; n < D_STATE; ++n) {
      float dA = __expf(dtv * A[n]);
      float Bn = __shfl(bc, n, 32);
      float Cn = __shfl(bc, n + D_STATE, 32);
      h[n] = fmaf(dA, h[n], dtu * Bn);
      acc  = fmaf(h[n], Cn, acc);
    }
    float y = acc + uv * Dv;
    float z = xz[row * (2 * D_INNER) + D_INNER + e];
    y *= siluf(z);
    y_bf[row * D_INNER + e] = f2bf(y);
  }
}

// ---------------------------------------------------------------------------
extern "C" void kernel_launch(void* const* d_in, const int* in_sizes, int n_in,
                              void* d_out, int out_size, void* d_ws, size_t ws_size,
                              hipStream_t stream) {
  (void)in_sizes; (void)n_in; (void)out_size; (void)ws_size;
  const float* x_in   = (const float*)d_in[0];
  const float* norm_w = (const float*)d_in[1];
  const float* norm_b = (const float*)d_in[2];
  const float* inW    = (const float*)d_in[3];
  const float* convW  = (const float*)d_in[4];
  const float* convB  = (const float*)d_in[5];
  const float* xprojW = (const float*)d_in[6];
  const float* dtW    = (const float*)d_in[7];
  const float* dtB    = (const float*)d_in[8];
  const float* A_log  = (const float*)d_in[9];
  const float* skipD  = (const float*)d_in[10];
  const float* outW   = (const float*)d_in[11];
  const float* fnW    = (const float*)d_in[12];
  const float* fnB    = (const float*)d_in[13];

  char* p = (char*)d_ws;
  auto take = [&](size_t bytes) -> char* {
    char* q = p; p += (bytes + 255) & ~(size_t)255; return q;
  };
  float*  xws  = (float*) take((size_t)ROWS * D_MODEL * 4);
  float*  xz   = (float*) take((size_t)ROWS * 2 * D_INNER * 4);
  float*  u    = (float*) take((size_t)ROWS * D_INNER * 4);
  float*  dt   = (float*) take((size_t)ROWS * D_INNER * 4);
  float*  dbl  = (float*) take((size_t)ROWS * DBL_COLS * 4);
  bf16_t* h_bf = (bf16_t*)take((size_t)ROWS * D_MODEL * 2);
  bf16_t* u_bf = (bf16_t*)take((size_t)ROWS * D_INNER * 2);
  bf16_t* y_bf = (bf16_t*)take((size_t)ROWS * D_INNER * 2);
  bf16_t* dtin = (bf16_t*)take((size_t)ROWS * DT_RANK * 2);
  bf16_t* w1t  = (bf16_t*)take((size_t)N_BLOCKS * 2 * D_INNER * D_MODEL * 2);
  bf16_t* w2t  = (bf16_t*)take((size_t)N_BLOCKS * DBL_COLS * D_INNER * 2);
  bf16_t* w3t  = (bf16_t*)take((size_t)N_BLOCKS * D_INNER * DT_RANK * 2);
  bf16_t* w4t  = (bf16_t*)take((size_t)N_BLOCKS * D_MODEL * D_INNER * 2);

  hipMemcpyAsync(xws, x_in, sizeof(float) * ROWS * D_MODEL,
                 hipMemcpyDeviceToDevice, stream);

  dim3 tgrid(1024, N_BLOCKS);
  transpose_to_bf16<<<tgrid, 256, 0, stream>>>(inW,    w1t, D_MODEL, 2 * D_INNER);
  transpose_to_bf16<<<tgrid, 256, 0, stream>>>(xprojW, w2t, D_INNER, DBL_COLS);
  transpose_to_bf16<<<tgrid, 256, 0, stream>>>(dtW,    w3t, DT_RANK, D_INNER);
  transpose_to_bf16<<<tgrid, 256, 0, stream>>>(outW,   w4t, D_INNER, D_MODEL);

  // 64x32 register blocks: tiles = (M/64) * (N/32), 8 waves per 256-thr block
  const int g_inproj = (ROWS / 64) * ((2 * D_INNER) / 32) / 8;  // 512
  const int g_xproj  = ((ROWS / 16) * (DBL_COLS / 16) + 7) / 8; // 96 (1x1 tiles)
  const int g_dtproj = (ROWS / 64) * (D_INNER / 32) / 8;        // 256
  const int g_outprj = (ROWS / 64) * (D_MODEL / 32) / 8;        // 128

  for (int i = 0; i < N_BLOCKS; ++i) {
    layernorm_kernel<<<ROWS, 256, 0, stream>>>(
        xws, norm_w + (size_t)i * D_MODEL, norm_b + (size_t)i * D_MODEL,
        nullptr, h_bf, D_MODEL);

    gemm_bf16_wmma<4, 2><<<g_inproj, 256, 0, stream>>>(
        h_bf, D_MODEL, w1t + (size_t)i * 2 * D_INNER * D_MODEL,
        xz, 2 * D_INNER, ROWS, 2 * D_INNER, D_MODEL, 0, nullptr);

    conv_silu_kernel<<<(BATCH * SEQLEN * D_INNER) / 256, 256, 0, stream>>>(
        xz, convW + (size_t)i * D_INNER * D_CONV, convB + (size_t)i * D_INNER,
        u, u_bf);

    gemm_bf16_wmma<1, 1><<<g_xproj, 256, 0, stream>>>(
        u_bf, D_INNER, w2t + (size_t)i * DBL_COLS * D_INNER,
        dbl, DBL_COLS, ROWS, DBL_COLS, D_INNER, 0, nullptr);

    cvt_bf16_strided<<<(ROWS * DT_RANK) / 256, 256, 0, stream>>>(
        dbl, dtin, ROWS, DT_RANK, DBL_COLS);

    gemm_bf16_wmma<4, 2><<<g_dtproj, 256, 0, stream>>>(
        dtin, DT_RANK, w3t + (size_t)i * D_INNER * DT_RANK,
        dt, D_INNER, ROWS, D_INNER, DT_RANK, 2, dtB + (size_t)i * D_INNER);

    scan_kernel<<<(BATCH * D_INNER) / 256, 256, 0, stream>>>(
        dt, u, dbl, xz, A_log + (size_t)i * D_INNER * D_STATE,
        skipD + (size_t)i * D_INNER, y_bf);

    gemm_bf16_wmma<4, 2><<<g_outprj, 256, 0, stream>>>(
        y_bf, D_INNER, w4t + (size_t)i * D_MODEL * D_INNER,
        xws, D_MODEL, ROWS, D_MODEL, D_INNER, 1, nullptr);
  }

  layernorm_kernel<<<ROWS, 256, 0, stream>>>(
      xws, fnW, fnB, (float*)d_out, nullptr, D_MODEL);
}
